// GCN_82351702934075
// MI455X (gfx1250) — compile-verified, matrix operand order checked by reading, not run on hardware
//
#include <hip/hip_runtime.h>

typedef __attribute__((ext_vector_type(16))) _Float16 v16h;
typedef __attribute__((ext_vector_type(8)))  _Float16 v8h;
typedef __attribute__((ext_vector_type(8)))  float    v8f;

#define EMB 64
#define HID 128
#define ACT 8

// ---------------- GCN-mean path (bandwidth/atomic bound) ----------------

__global__ void k_init(float* deg, float* g, int n_nodes) {
  int i = blockIdx.x * blockDim.x + threadIdx.x;
  if (i < n_nodes) deg[i] = 1.0f;   // self-loop weight
  if (i < EMB) g[i] = 0.0f;
}

__global__ void k_deg(const int* __restrict__ dst, const float* __restrict__ w,
                      float* deg, int n_edges) {
  for (int e = blockIdx.x * blockDim.x + threadIdx.x; e < n_edges;
       e += gridDim.x * blockDim.x)
    atomicAdd(&deg[dst[e]], w[e]);
}

__global__ void k_dinv(const float* __restrict__ deg, float* dinv, float* c, int n_nodes) {
  int i = blockIdx.x * blockDim.x + threadIdx.x;
  if (i < n_nodes) {
    float d = deg[i];
    float r = d > 0.f ? rsqrtf(d) : 0.f;
    dinv[i] = r;
    c[i] = r * r;                   // self-loop contribution to per-src coefficient
  }
}

__global__ void k_coeff(const int* __restrict__ src, const int* __restrict__ dst,
                        const float* __restrict__ w, const float* __restrict__ dinv,
                        float* c, int n_edges) {
  for (int e = blockIdx.x * blockDim.x + threadIdx.x; e < n_edges;
       e += gridDim.x * blockDim.x) {
    int s = src[e], d = dst[e];
    atomicAdd(&c[s], dinv[s] * w[e] * dinv[d]);
  }
}

// g[j] = sum_s c[s] * emb[s][j]   (j in 0..63)
__global__ void k_reduce_g(const float* __restrict__ c, const float* __restrict__ emb,
                           float* g, int n_nodes) {
  __shared__ float part[256];
  const int j    = threadIdx.x & 63;
  const int slot = threadIdx.x >> 6;  // 0..3
  float acc = 0.f;
  for (int i = blockIdx.x * 4 + slot; i < n_nodes; i += gridDim.x * 4)
    acc += c[i] * emb[i * EMB + j];
  part[threadIdx.x] = acc;
  __syncthreads();
  if (threadIdx.x < 64) {
    float s = part[threadIdx.x] + part[64 + threadIdx.x] +
              part[128 + threadIdx.x] + part[192 + threadIdx.x];
    atomicAdd(&g[threadIdx.x], s);
  }
}

__global__ void k_x2m(const float* __restrict__ g, const float* __restrict__ Wg,
                      const float* __restrict__ bg, float* x2m, int n_nodes) {
  int j = threadIdx.x;
  if (j < HID) {
    float acc = 0.f;
#pragma unroll
    for (int k = 0; k < EMB; ++k) acc += g[k] * Wg[k * HID + j];
    x2m[j] = acc / (float)n_nodes + bg[j];
  }
}

// ---------------- Fused WMMA MLP head ----------------

__device__ __forceinline__ v8f wmma16(v16h a, v16h b, v8f c) {
  return __builtin_amdgcn_wmma_f32_16x16x32_f16(
      /*neg_a=*/false, a, /*neg_b=*/false, b,
      /*c_mod=*/(short)0, c, /*reuse_a=*/false, /*reuse_b=*/false);
}

// A fragment (16xK tile, row-major f16 in LDS), per ISA 16-bit A layout
__device__ __forceinline__ v16h ldsA(const _Float16* s, int ld, int kbase, int lane) {
  const int r  = lane & 15;
  const int kb = kbase + ((lane & 16) ? 8 : 0);
  v8h lo = *(const v8h*)(s + r * ld + kb);
  v8h hi = *(const v8h*)(s + r * ld + kb + 16);
  v16h a;
#pragma unroll
  for (int i = 0; i < 8; ++i) { a[i] = lo[i]; a[i + 8] = hi[i]; }
  return a;
}

// B fragment from row-major f32 weight W[K][ldb], column block ncol0
__device__ __forceinline__ v16h gB(const float* __restrict__ W, int ldb, int ncol0,
                                   int kbase, int lane) {
  const int col = ncol0 + (lane & 15);
  const int kb  = kbase + ((lane & 16) ? 8 : 0);
  v16h b;
#pragma unroll
  for (int i = 0; i < 8; ++i) b[i] = (_Float16)W[(kb + i) * ldb + col];
#pragma unroll
  for (int i = 0; i < 8; ++i) b[i + 8] = (_Float16)W[(kb + 16 + i) * ldb + col];
  return b;
}

// store D (16x16 f32 acc) to f16 LDS with bias + relu, per ISA C/D layout
__device__ __forceinline__ void storeD_relu_h(_Float16* s, int ld, int ncol0, int lane,
                                              v8f acc, const float* __restrict__ bias) {
  const int n  = ncol0 + (lane & 15);
  const int mb = (lane & 16) ? 8 : 0;
  const float bb = bias[n];
#pragma unroll
  for (int r = 0; r < 8; ++r) {
    float v = acc[r] + bb;
    s[(mb + r) * ld + n] = (_Float16)(v > 0.f ? v : 0.f);
  }
}

__global__ void __launch_bounds__(256) k_mlp(
    const int* __restrict__ state, const float* __restrict__ emb,
    const float* __restrict__ x2m,
    const float* __restrict__ W1, const float* __restrict__ b1,
    const float* __restrict__ W2, const float* __restrict__ b2,
    const float* __restrict__ W3, const float* __restrict__ b3,
    const float* __restrict__ Wv, const float* __restrict__ bv,
    const float* __restrict__ Wa, const float* __restrict__ ba,
    float* __restrict__ out) {
  __shared__ __align__(16) _Float16 sA[16 * 192];   // xcat tile (f16)
  __shared__ __align__(16) _Float16 sH[16 * HID];   // hidden tile (f16)
  __shared__ __align__(16) float    sH3[16 * HID];  // final hidden (f32)

  const int tid     = threadIdx.x;
  const int rowbase = blockIdx.x * 16;

  // stage xcat = [emb[state[row]], x2mean] as f16
  for (int idx = tid; idx < 16 * 192; idx += 256) {
    int r = idx / 192, k = idx - r * 192;
    float v = (k < EMB) ? emb[(long)state[rowbase + r] * EMB + k] : x2m[k - EMB];
    sA[idx] = (_Float16)v;
  }
  __syncthreads();

  const int wave  = tid >> 5;   // 0..7 -> N tile
  const int lane  = tid & 31;
  const int ncol0 = wave * 16;

  // layer 1: (16x192) @ (192x128), K unrolled: 6 WMMA
  v8f acc1 = {};
#pragma unroll
  for (int kt = 0; kt < 6; ++kt)
    acc1 = wmma16(ldsA(sA, 192, kt * 32, lane), gB(W1, HID, ncol0, kt * 32, lane), acc1);
  storeD_relu_h(sH, HID, ncol0, lane, acc1, b1);
  __syncthreads();

  // layer 2: (16x128) @ (128x128): 4 WMMA
  v8f acc2 = {};
#pragma unroll
  for (int kt = 0; kt < 4; ++kt)
    acc2 = wmma16(ldsA(sH, HID, kt * 32, lane), gB(W2, HID, ncol0, kt * 32, lane), acc2);
  __syncthreads();  // all reads of sH complete before overwrite
  storeD_relu_h(sH, HID, ncol0, lane, acc2, b2);
  __syncthreads();

  // layer 3: (16x128) @ (128x128): 4 WMMA, store f32
  v8f acc3 = {};
#pragma unroll
  for (int kt = 0; kt < 4; ++kt)
    acc3 = wmma16(ldsA(sH, HID, kt * 32, lane), gB(W3, HID, ncol0, kt * 32, lane), acc3);
  {
    const int n  = ncol0 + (lane & 15);
    const int mb = (lane & 16) ? 8 : 0;
    const float bb = b3[n];
#pragma unroll
    for (int r = 0; r < 8; ++r) {
      float v = acc3[r] + bb;
      sH3[(mb + r) * HID + n] = v > 0.f ? v : 0.f;
    }
  }
  __syncthreads();

  // dueling heads: value (128->1), adv (128->8), out = v + adv - mean(adv)
  if (tid < 16) {
    const float* hrow = &sH3[tid * HID];
    float val = bv[0];
    float adv[ACT];
#pragma unroll
    for (int a = 0; a < ACT; ++a) adv[a] = ba[a];
    for (int k = 0; k < HID; ++k) {
      float hv = hrow[k];
      val += hv * Wv[k];
#pragma unroll
      for (int a = 0; a < ACT; ++a) adv[a] += hv * Wa[k * ACT + a];
    }
    float m = 0.f;
#pragma unroll
    for (int a = 0; a < ACT; ++a) m += adv[a];
    m *= (1.0f / ACT);
    float* o = out + (long)(rowbase + tid) * ACT;
#pragma unroll
    for (int a = 0; a < ACT; ++a) o[a] = val + adv[a] - m;
  }
}

// ---------------- launch ----------------

extern "C" void kernel_launch(void* const* d_in, const int* in_sizes, int n_in,
                              void* d_out, int out_size, void* d_ws, size_t ws_size,
                              hipStream_t stream) {
  const int*   state = (const int*)d_in[0];
  const int*   ei    = (const int*)d_in[2];
  const float* ew    = (const float*)d_in[3];
  const float* emb   = (const float*)d_in[4];
  const float* Wg    = (const float*)d_in[5];
  const float* bg    = (const float*)d_in[6];
  const float* W1    = (const float*)d_in[7];
  const float* b1    = (const float*)d_in[8];
  const float* W2    = (const float*)d_in[9];
  const float* b2    = (const float*)d_in[10];
  const float* W3    = (const float*)d_in[11];
  const float* b3    = (const float*)d_in[12];
  const float* Wv    = (const float*)d_in[13];
  const float* bv    = (const float*)d_in[14];
  const float* Wa    = (const float*)d_in[15];
  const float* ba    = (const float*)d_in[16];
  float* out = (float*)d_out;

  const int batch   = in_sizes[0];
  const int n_edges = in_sizes[2] / 2;
  const int n_nodes = in_sizes[4] / EMB;
  const int* src = ei;
  const int* dst = ei + n_edges;

  float* deg  = (float*)d_ws;
  float* dinv = deg + n_nodes;
  float* c    = dinv + n_nodes;
  float* g    = c + n_nodes;   // 64 floats
  float* x2m  = g + EMB;       // 128 floats

  int nbN = (n_nodes + 255) / 256;
  int nbE = (n_edges + 255) / 256;
  if (nbE > 2048) nbE = 2048;

  k_init  <<<nbN, 256, 0, stream>>>(deg, g, n_nodes);
  k_deg   <<<nbE, 256, 0, stream>>>(dst, ew, deg, n_edges);
  k_dinv  <<<nbN, 256, 0, stream>>>(deg, dinv, c, n_nodes);
  k_coeff <<<nbE, 256, 0, stream>>>(src, dst, ew, dinv, c, n_edges);
  k_reduce_g<<<256, 256, 0, stream>>>(c, emb, g, n_nodes);
  k_x2m   <<<1, 128, 0, stream>>>(g, Wg, bg, x2m, n_nodes);
  k_mlp   <<<batch / 16, 256, 0, stream>>>(state, emb, x2m, W1, b1, W2, b2,
                                           W3, b3, Wv, bv, Wa, ba, out);
}
